// ContrastiveLoss_55482387529738
// MI455X (gfx1250) — matrix operand order repaired and checked by reading, not compiled, not run on hardware
//
#include <hip/hip_runtime.h>

typedef __attribute__((ext_vector_type(16))) _Float16 v16h;
typedef __attribute__((ext_vector_type(8)))  _Float16 v8h;
typedef __attribute__((ext_vector_type(2)))  _Float16 v2h;
typedef __attribute__((ext_vector_type(8)))  float    v8f;

#define NROWS 16384
#define DIM   64
#define SPLIT 8
#define JT_PER_WAVE 16          // (NROWS/16) j-tiles / SPLIT / 8 waves = 1024/64

// invT = 1/0.07 ; SCALE = invT * log2(e) (fold exp -> exp2 native v_exp_f32)
__device__ __constant__ const float kInvT  = 14.285714285714286f;
__device__ __constant__ const float kScale = 14.285714285714286f * 1.4426950408889634f;
__device__ __constant__ const float kLn2   = 0.6931471805599453f;

__device__ __forceinline__ v16h pack16(v8h lo, v8h hi) {
    return __builtin_shufflevector(lo, hi, 0,1,2,3,4,5,6,7,8,9,10,11,12,13,14,15);
}

// ---------------- Phase 1: fp32 row-normalize -> f16, exact fp32 diagonal ----
extern "C" __global__ __launch_bounds__(256)
void cl_normalize(const float* __restrict__ A, const float* __restrict__ B,
                  _Float16* __restrict__ Ah, _Float16* __restrict__ Bh,
                  float* __restrict__ diag)
{
    const int wave = threadIdx.x >> 5;
    const int lane = threadIdx.x & 31;
    const int row  = blockIdx.x * 8 + wave;

    const float2 a = *(const float2*)(A + row * DIM + lane * 2);
    const float2 b = *(const float2*)(B + row * DIM + lane * 2);

    float sa = a.x * a.x + a.y * a.y;
    float sb = b.x * b.x + b.y * b.y;
#pragma unroll
    for (int m = 16; m >= 1; m >>= 1) {
        sa += __shfl_xor(sa, m, 32);
        sb += __shfl_xor(sb, m, 32);
    }
    const float ra = __builtin_amdgcn_rsqf(sa);
    const float rb = __builtin_amdgcn_rsqf(sb);

    const float an0 = a.x * ra, an1 = a.y * ra;
    const float bn0 = b.x * rb, bn1 = b.y * rb;

    // exact fp32 diagonal sim (a_i . b_i) / T
    float d = an0 * bn0 + an1 * bn1;
#pragma unroll
    for (int m = 16; m >= 1; m >>= 1) d += __shfl_xor(d, m, 32);

    v2h ah; ah[0] = (_Float16)an0; ah[1] = (_Float16)an1;
    v2h bh; bh[0] = (_Float16)bn0; bh[1] = (_Float16)bn1;
    *(v2h*)(Ah + row * DIM + lane * 2) = ah;
    *(v2h*)(Bh + row * DIM + lane * 2) = bh;

    if (lane == 0) diag[row] = d * kInvT;
}

// ---------------- Phase 2: WMMA 16x16 sim tiles fused with shifted exp-sum ---
// Grid: (NROWS/16) * SPLIT blocks, 256 threads (8 waves).
// Each wave: one 16-row A tile resident in registers, sweeps JT_PER_WAVE B tiles.
extern "C" __global__ __launch_bounds__(256)
void cl_simtile(const _Float16* __restrict__ Ah, const _Float16* __restrict__ Bh,
                float* __restrict__ partial /* [SPLIT][NROWS] */)
{
    const int rt    = blockIdx.x / SPLIT;     // row tile 0..1023
    const int split = blockIdx.x % SPLIT;
    // wave index is wave-uniform: force it scalar so the jt loop is a plain
    // SALU counter (no per-iteration v_cmp / exec-mask manipulation).
    const int wave  = __builtin_amdgcn_readfirstlane(threadIdx.x >> 5);
    const int lane  = threadIdx.x & 31;
    const int r     = lane & 15;              // row / column within tile
    const int h     = lane >> 4;              // half-wave

    // A tile (16x64 f16) in ISA A-layout: half h holds K in {8h..8h+7} U {16+8h..16+8h+7}
    const _Float16* arow = Ah + (size_t)(rt * 16 + r) * DIM;
    const v16h a0 = pack16(*(const v8h*)(arow +       8 * h),
                           *(const v8h*)(arow + 16 +  8 * h));
    const v16h a1 = pack16(*(const v8h*)(arow + 32 +  8 * h),
                           *(const v8h*)(arow + 48 +  8 * h));

    float rs[8];
#pragma unroll
    for (int i = 0; i < 8; ++i) rs[i] = 0.f;

    const int jt0 = (split * 8 + wave) * JT_PER_WAVE;
    // Per-lane B base: row (= sim column) r of each 16-row tile.
    const _Float16* bbase = Bh + (size_t)(jt0 * 16 + r) * DIM;

#pragma unroll 2
    for (int j = 0; j < JT_PER_WAVE; ++j) {
        // B tile: sim = A . B^T, so B columns are b-rows; B-layout matches A:
        // half h holds K in {8h..8h+7} U {16+8h..16+8h+7} for lane-column r.
        const _Float16* brow = bbase + (size_t)j * (16 * DIM);
        const v16h b0 = pack16(*(const v8h*)(brow +       8 * h),
                               *(const v8h*)(brow + 16 +  8 * h));
        const v16h b1 = pack16(*(const v8h*)(brow + 32 +  8 * h),
                               *(const v8h*)(brow + 48 +  8 * h));

        v8f c = {};
        c = __builtin_amdgcn_wmma_f32_16x16x32_f16(false, a0, false, b0,
                                                   (short)0, c, false, false);
        c = __builtin_amdgcn_wmma_f32_16x16x32_f16(false, a1, false, b1,
                                                   (short)0, c, false, false);

        // c[rr] = dot(row 8h+rr, col r). Shifted exp2: 2^(dot*S - S), S = invT*log2e.
#pragma unroll
        for (int rr = 0; rr < 8; ++rr)
            rs[rr] += __builtin_amdgcn_exp2f(c[rr] * kScale - kScale);
    }

    // Reduce over the 16 lane-columns within each half (rows differ across halves)
#pragma unroll
    for (int rr = 0; rr < 8; ++rr) {
#pragma unroll
        for (int m = 8; m >= 1; m >>= 1) rs[rr] += __shfl_xor(rs[rr], m, 32);
    }

    __shared__ float lds[8 * 16];   // [wave][row-in-tile]
    if (r == 0) {
#pragma unroll
        for (int rr = 0; rr < 8; ++rr) lds[wave * 16 + h * 8 + rr] = rs[rr];
    }
    __syncthreads();

    if (threadIdx.x < 16) {
        float s = 0.f;
#pragma unroll
        for (int w = 0; w < 8; ++w) s += lds[w * 16 + threadIdx.x];
        partial[(size_t)split * NROWS + rt * 16 + threadIdx.x] = s;
    }
}

// ---------------- Phase 3: per-row loss + deterministic mean -----------------
extern "C" __global__ __launch_bounds__(256)
void cl_finalize(const float* __restrict__ partial, const float* __restrict__ diag,
                 float* __restrict__ out)
{
    float acc = 0.f;
    for (int row = threadIdx.x; row < NROWS; row += 256) {
        float s = 0.f;
#pragma unroll
        for (int k = 0; k < SPLIT; ++k) s += partial[(size_t)k * NROWS + row];
        // lse = (log2(s) + SHIFT) * ln2 ; SHIFT == kScale
        acc += (__builtin_amdgcn_logf(s) + kScale) * kLn2 - diag[row];
    }
    __shared__ float red[256];
    red[threadIdx.x] = acc;
    __syncthreads();
#pragma unroll
    for (int st = 128; st > 0; st >>= 1) {
        if (threadIdx.x < st) red[threadIdx.x] += red[threadIdx.x + st];
        __syncthreads();
    }
    if (threadIdx.x == 0) out[0] = red[0] * (1.0f / NROWS);
}

// ---------------- Host launch ------------------------------------------------
extern "C" void kernel_launch(void* const* d_in, const int* in_sizes, int n_in,
                              void* d_out, int out_size, void* d_ws, size_t ws_size,
                              hipStream_t stream)
{
    const float* A = (const float*)d_in[0];
    const float* B = (const float*)d_in[1];

    char* ws = (char*)d_ws;
    _Float16* Ah  = (_Float16*)ws;                                  // 2 MB
    _Float16* Bh  = (_Float16*)(ws + (size_t)2 * 1024 * 1024);      // 2 MB
    float*    dg  = (float*)(ws + (size_t)4 * 1024 * 1024);         // 64 KB
    float*    prt = (float*)(ws + (size_t)4 * 1024 * 1024 + 65536); // 512 KB

    cl_normalize<<<NROWS / 8, 256, 0, stream>>>(A, B, Ah, Bh, dg);
    cl_simtile<<<(NROWS / 16) * SPLIT, 256, 0, stream>>>(Ah, Bh, prt);
    cl_finalize<<<1, 256, 0, stream>>>(prt, dg, (float*)d_out);
}
